// GALE_FA_74912819577588
// MI455X (gfx1250) — matrix operand order repaired and checked by reading, not compiled, not run on hardware
//
#include <hip/hip_runtime.h>

typedef __bf16 bf16;
typedef __attribute__((ext_vector_type(16))) __bf16 v16bf;
typedef __attribute__((ext_vector_type(8)))  __bf16 v8bf;
typedef __attribute__((ext_vector_type(8)))  float  v8f;
typedef __attribute__((ext_vector_type(4)))  int    i4;

#define B_   2
#define N_   16384
#define DIM_ 512
#define H_   8
#define D_   64
#define M_   64
#define SC_  64

#if defined(__has_builtin)
#if __has_builtin(__builtin_amdgcn_global_load_async_to_lds_b128)
#define HAVE_ASYNC 1
#endif
#if __has_builtin(__builtin_amdgcn_s_wait_asynccnt)
#define HAVE_WAIT_BUILTIN 1
#endif
#endif
#ifndef HAVE_ASYNC
#define HAVE_ASYNC 0
#endif
#ifndef HAVE_WAIT_BUILTIN
#define HAVE_WAIT_BUILTIN 0
#endif

typedef __attribute__((address_space(1))) i4 gi4;  // global 16B vector
typedef __attribute__((address_space(3))) i4 li4;  // LDS 16B vector

// 16-byte global -> LDS copy (per lane). Async (ASYNCcnt) when available.
__device__ __forceinline__ void async_cp16(const bf16* g, bf16* l) {
#if HAVE_ASYNC
  __builtin_amdgcn_global_load_async_to_lds_b128((gi4*)g, (li4*)l, 0, 0);
#else
  *(uint4*)l = *(const uint4*)g;
#endif
}
__device__ __forceinline__ void wait_async0() {
#if HAVE_ASYNC
#if HAVE_WAIT_BUILTIN
  __builtin_amdgcn_s_wait_asynccnt(0);
#else
  asm volatile("s_wait_asynccnt 0x0" ::: "memory");
#endif
#endif
}
__device__ __forceinline__ void wait_async6() {
#if HAVE_ASYNC
#if HAVE_WAIT_BUILTIN
  __builtin_amdgcn_s_wait_asynccnt(6);
#else
  asm volatile("s_wait_asynccnt 0x6" ::: "memory");
#endif
#endif
}

__device__ __forceinline__ v8f vzero8() {
  v8f z;
#pragma unroll
  for (int i = 0; i < 8; ++i) z[i] = 0.f;
  return z;
}

__device__ __forceinline__ v8f wmma_bf16(v16bf a, v16bf b, v8f c) {
  // D = A(16x32) * B(32x16) + C, f32 accumulate
  return __builtin_amdgcn_wmma_f32_16x16x32_bf16(false, a, false, b, (short)0, c, false, false);
}

// A fragment: 16x32 bf16, row-major [m][k], ld = row stride in elements.
// lane<16: row=lane, K=0..15 ; lane>=16: row=lane-16, K=16..31
__device__ __forceinline__ v16bf ldsA(const bf16* base, int ld, int m0, int k0, int lane) {
  int r  = m0 + (lane & 15);
  int kq = k0 + ((lane >> 4) << 4);
  const bf16* p = base + r * ld + kq;
  v16bf x;
  *(v8bf*)&x       = *(const v8bf*)p;         // K = kq .. kq+7
  *((v8bf*)&x + 1) = *(const v8bf*)(p + 8);   // K = kq+8 .. kq+15
  return x;
}

// B fragment: 32x16 bf16 staged TRANSPOSED in LDS as B^T [n][k], ld = K stride.
// lane<16: col=lane, K=0..15 ; lane>=16: col=lane-16, K=16..31
__device__ __forceinline__ v16bf ldsB(const bf16* baseT, int ld, int n0, int k0, int lane) {
  int c  = n0 + (lane & 15);
  int kh = k0 + ((lane >> 4) << 4);
  const bf16* p = baseT + c * ld + kh;
  v16bf x;
  *(v8bf*)&x       = *(const v8bf*)p;
  *((v8bf*)&x + 1) = *(const v8bf*)(p + 8);
  return x;
}

__device__ __forceinline__ void atomicMaxF(float* addr, float val) {
  unsigned int* ia = (unsigned int*)addr;
  unsigned int old = *ia;
  while (__uint_as_float(old) < val) {
    unsigned int assumed = old;
    old = atomicCAS(ia, assumed, __float_as_uint(val));
    if (old == assumed) break;
  }
}

// ---------------------------------------------------------------- fp32 -> bf16
__global__ void k_f2bf(const float* __restrict__ in, bf16* __restrict__ out, int n) {
  int i = blockIdx.x * 256 + threadIdx.x;
  if (i < n) out[i] = (bf16)in[i];
}

// ------------------------------------------- context projections (tiny GEMMs)
__global__ void k_ctx(const float* __restrict__ ctx,
                      const float* __restrict__ Wck, const float* __restrict__ bck,
                      const float* __restrict__ Wcv, const float* __restrict__ bcv,
                      bf16* __restrict__ kc, bf16* __restrict__ vcT) {
  int i = blockIdx.x * 256 + threadIdx.x;
  if (i >= B_ * H_ * SC_ * D_) return;
  int d = i & 63, s = (i >> 6) & 63, bh = i >> 12;
  const float* cp = ctx + ((size_t)bh * SC_ + s) * 64;
  const float* wk = Wck + d * 64;
  const float* wv = Wcv + d * 64;
  float ak = bck[d], av = bcv[d];
#pragma unroll 8
  for (int e = 0; e < 64; ++e) { float c = cp[e]; ak += c * wk[e]; av += c * wv[e]; }
  kc [((size_t)bh * SC_ + s) * D_ + d]  = (bf16)ak;
  vcT[((size_t)bh * D_  + d) * SC_ + s] = (bf16)av;
}

// -------------------- x -> x_mid (per head) -> k, v, q  (all bf16 WMMA GEMMs)
__global__ void __launch_bounds__(128)
k_proj_qkv(const float* __restrict__ x, const bf16* __restrict__ Wx, const float* __restrict__ bx,
           const bf16* __restrict__ Wk, const float* __restrict__ bk,
           const bf16* __restrict__ Wv, const float* __restrict__ bv,
           const bf16* __restrict__ Wcq, const float* __restrict__ bcq,
           bf16* __restrict__ kO, bf16* __restrict__ vO, bf16* __restrict__ qO) {
  __shared__ bf16 As[128 * 32];   // x tile, [token][k]
  __shared__ bf16 Bs[64 * 32];    // Wx head rows = B^T
  __shared__ bf16 Xm[128 * 64];   // x_mid tile (bf16)
  __shared__ bf16 Ws[64 * 64];    // 64x64 head weight = B^T

  const int tid = threadIdx.x, lane = tid & 31, w = tid >> 5;
  const int t0 = blockIdx.x * 128;
  const int h  = blockIdx.y;

  v8f acc[2][4];
#pragma unroll
  for (int rt = 0; rt < 2; ++rt)
#pragma unroll
    for (int ct = 0; ct < 4; ++ct) acc[rt][ct] = vzero8();

  for (int k0 = 0; k0 < DIM_; k0 += 32) {
    // B^T tile from Wx (rows h*64..h*64+63) -- async to LDS
    {
      int r = tid >> 1, cb = (tid & 1) * 16;
      const bf16* gb = Wx + (size_t)(h * 64 + r) * DIM_ + k0 + cb;
      bf16* lb = Bs + r * 32 + cb;
      async_cp16(gb, lb);
      async_cp16(gb + 8, lb + 8);
    }
    // A tile: one row per thread, f32 -> bf16 (needs conversion, stays sync)
    {
      const float4* xv = (const float4*)(x + (size_t)(t0 + tid) * DIM_ + k0);
      __builtin_prefetch(xv + 8, 0, 1);
      bf16* ap = As + tid * 32;
#pragma unroll
      for (int i = 0; i < 8; ++i) {
        float4 f = xv[i];
        ap[4 * i + 0] = (bf16)f.x; ap[4 * i + 1] = (bf16)f.y;
        ap[4 * i + 2] = (bf16)f.z; ap[4 * i + 3] = (bf16)f.w;
      }
    }
    wait_async0();
    __syncthreads();
    v16bf bb[4];
#pragma unroll
    for (int ct = 0; ct < 4; ++ct) bb[ct] = ldsB(Bs, 32, ct * 16, 0, lane);
#pragma unroll
    for (int rt = 0; rt < 2; ++rt) {
      v16bf a = ldsA(As, 32, w * 32 + rt * 16, 0, lane);
#pragma unroll
      for (int ct = 0; ct < 4; ++ct) acc[rt][ct] = wmma_bf16(a, bb[ct], acc[rt][ct]);
    }
    __syncthreads();
  }

  // bias + store x_mid tile to LDS (wave-private rows)
#pragma unroll
  for (int rt = 0; rt < 2; ++rt)
#pragma unroll
    for (int ct = 0; ct < 4; ++ct) {
      int col = ct * 16 + (lane & 15);
      float bc = bx[h * 64 + col];
#pragma unroll
      for (int e = 0; e < 8; ++e) {
        int row = w * 32 + rt * 16 + ((lane >> 4) << 3) + e;
        Xm[row * 64 + col] = (bf16)(acc[rt][ct][e] + bc);
      }
    }
  __syncthreads();

  const bf16* Wp[3]  = {Wk, Wv, Wcq};
  const float* bp[3] = {bk, bv, bcq};
  bf16* op[3]        = {kO, vO, qO};
#pragma unroll 1
  for (int pi = 0; pi < 3; ++pi) {
    for (int i = tid; i < 512; i += 128) async_cp16(Wp[pi] + 8 * i, Ws + 8 * i);
    wait_async0();
    __syncthreads();
    v8f a2[2][4];
#pragma unroll
    for (int rt = 0; rt < 2; ++rt)
#pragma unroll
      for (int ct = 0; ct < 4; ++ct) a2[rt][ct] = vzero8();
#pragma unroll
    for (int k0 = 0; k0 < 64; k0 += 32) {
      v16bf bb[4];
#pragma unroll
      for (int ct = 0; ct < 4; ++ct) bb[ct] = ldsB(Ws, 64, ct * 16, k0, lane);
#pragma unroll
      for (int rt = 0; rt < 2; ++rt) {
        v16bf a = ldsA(Xm, 64, w * 32 + rt * 16, k0, lane);
#pragma unroll
        for (int ct = 0; ct < 4; ++ct) a2[rt][ct] = wmma_bf16(a, bb[ct], a2[rt][ct]);
      }
    }
#pragma unroll
    for (int rt = 0; rt < 2; ++rt)
#pragma unroll
      for (int ct = 0; ct < 4; ++ct) {
        int col = ct * 16 + (lane & 15);
        float bc = bp[pi][col];
#pragma unroll
        for (int e = 0; e < 8; ++e) {
          int row = w * 32 + rt * 16 + ((lane >> 4) << 3) + e;
          int t   = t0 + row;
          int b   = t / N_, n = t - b * N_;
          op[pi][(((size_t)(b * H_ + h)) * N_ + n) * D_ + col] = (bf16)(a2[rt][ct][e] + bc);
        }
      }
    __syncthreads();
  }
}

// --------------------------------------------------------- encode stage setup
__global__ void k_einit(float* lat, float* rmax, float* lsum) {
  int i = blockIdx.x * 256 + threadIdx.x;
  if (i < B_ * H_ * M_ * D_) lat[i] = 0.f;
  else if (i < B_ * H_ * M_ * D_ + B_ * H_ * M_) rmax[i - B_ * H_ * M_ * D_] = -3.0e38f;
  else if (i < B_ * H_ * M_ * D_ + 2 * B_ * H_ * M_) lsum[i - B_ * H_ * M_ * D_ - B_ * H_ * M_] = 0.f;
}

// ------------------------------- encode pass 1: rowmax of Z = qg @ k^T (WMMA)
__global__ void __launch_bounds__(128)
k_encmax(const bf16* __restrict__ qg, const bf16* __restrict__ kbf, float* __restrict__ rmax) {
  __shared__ bf16 Qg[M_ * D_];    // A: [m][d]
  __shared__ bf16 Kt[128 * D_];   // B^T: [n][d]
  const int tid = threadIdx.x, lane = tid & 31, w = tid >> 5;
  const int bh = blockIdx.x, h = bh % H_;
  const int n0 = blockIdx.y * 128;
  {
    const bf16* gq = qg + (size_t)h * M_ * D_;
    for (int i = tid; i < 512; i += 128) async_cp16(gq + 8 * i, Qg + 8 * i);
    const bf16* gk = kbf + ((size_t)bh * N_ + n0) * D_;
    for (int i = tid; i < 1024; i += 128) async_cp16(gk + 8 * i, Kt + 8 * i);
  }
  wait_async0();
  __syncthreads();
  v8f acc[4][2];
#pragma unroll
  for (int rt = 0; rt < 4; ++rt) { acc[rt][0] = vzero8(); acc[rt][1] = vzero8(); }
#pragma unroll
  for (int k0 = 0; k0 < 64; k0 += 32) {
    v16bf bb[2];
#pragma unroll
    for (int ct = 0; ct < 2; ++ct) bb[ct] = ldsB(Kt, D_, w * 32 + ct * 16, k0, lane);
#pragma unroll
    for (int rt = 0; rt < 4; ++rt) {
      v16bf a = ldsA(Qg, D_, rt * 16, k0, lane);
#pragma unroll
      for (int ct = 0; ct < 2; ++ct) acc[rt][ct] = wmma_bf16(a, bb[ct], acc[rt][ct]);
    }
  }
#pragma unroll
  for (int rt = 0; rt < 4; ++rt)
#pragma unroll
    for (int e = 0; e < 8; ++e) {
      float mx = fmaxf(acc[rt][0][e], acc[rt][1][e]);
#pragma unroll
      for (int m = 8; m >= 1; m >>= 1) mx = fmaxf(mx, __shfl_xor(mx, m, 32));
      if ((lane & 15) == 0) {
        int row = rt * 16 + ((lane >> 4) << 3) + e;
        atomicMaxF(rmax + bh * M_ + row, mx);
      }
    }
}

// --------------- encode pass 2: P = exp(Z-max); lsum += P; latent += P@V (WMMA)
__global__ void __launch_bounds__(128)
k_encacc(const bf16* __restrict__ qg, const bf16* __restrict__ kbf, const bf16* __restrict__ vbf,
         const float* __restrict__ rmax, float* __restrict__ lsum, float* __restrict__ latent) {
  __shared__ bf16 Qg[M_ * D_];
  __shared__ bf16 Kt[128 * D_];
  __shared__ bf16 VtT[D_ * 128];  // B^T: [d][n]
  __shared__ bf16 Ps[M_ * 128];   // A:   [m][n]
  __shared__ float Rm[M_];
  const int tid = threadIdx.x, lane = tid & 31, w = tid >> 5;
  const int bh = blockIdx.x, h = bh % H_;
  const int n0 = blockIdx.y * 128;
  size_t kb = ((size_t)bh * N_ + n0) * D_;
  {
    const bf16* gq = qg + (size_t)h * M_ * D_;
    for (int i = tid; i < 512; i += 128) async_cp16(gq + 8 * i, Qg + 8 * i);
    const bf16* gk = kbf + kb;
    for (int i = tid; i < 1024; i += 128) async_cp16(gk + 8 * i, Kt + 8 * i);
    for (int i = tid; i < 8192; i += 128) {
      int nl = i >> 6, d = i & 63;
      VtT[d * 128 + nl] = vbf[kb + i];
    }
    if (tid < M_) Rm[tid] = rmax[bh * M_ + tid];
  }
  wait_async0();
  __syncthreads();
  v8f acc[4][2];
#pragma unroll
  for (int rt = 0; rt < 4; ++rt) { acc[rt][0] = vzero8(); acc[rt][1] = vzero8(); }
#pragma unroll
  for (int k0 = 0; k0 < 64; k0 += 32) {
    v16bf bb[2];
#pragma unroll
    for (int ct = 0; ct < 2; ++ct) bb[ct] = ldsB(Kt, D_, w * 32 + ct * 16, k0, lane);
#pragma unroll
    for (int rt = 0; rt < 4; ++rt) {
      v16bf a = ldsA(Qg, D_, rt * 16, k0, lane);
#pragma unroll
      for (int ct = 0; ct < 2; ++ct) acc[rt][ct] = wmma_bf16(a, bb[ct], acc[rt][ct]);
    }
  }
#pragma unroll
  for (int rt = 0; rt < 4; ++rt)
#pragma unroll
    for (int e = 0; e < 8; ++e) {
      int row = rt * 16 + ((lane >> 4) << 3) + e;
      float rm = Rm[row];
      float p0 = __expf(acc[rt][0][e] - rm);
      float p1 = __expf(acc[rt][1][e] - rm);
      float s  = p0 + p1;
#pragma unroll
      for (int m = 8; m >= 1; m >>= 1) s += __shfl_xor(s, m, 32);
      if ((lane & 15) == 0) atomicAdd(lsum + bh * M_ + row, s);
      int c = lane & 15;
      Ps[row * 128 + w * 32 + c]      = (bf16)p0;
      Ps[row * 128 + w * 32 + 16 + c] = (bf16)p1;
    }
  __syncthreads();
  v8f lat[4][4];
#pragma unroll
  for (int rt = 0; rt < 4; ++rt)
#pragma unroll
    for (int ct = 0; ct < 4; ++ct) lat[rt][ct] = vzero8();
  {
    v16bf bb[4];
#pragma unroll
    for (int ct = 0; ct < 4; ++ct) bb[ct] = ldsB(VtT, 128, ct * 16, w * 32, lane);
#pragma unroll
    for (int rt = 0; rt < 4; ++rt) {
      v16bf a = ldsA(Ps, 128, rt * 16, w * 32, lane);
#pragma unroll
      for (int ct = 0; ct < 4; ++ct) lat[rt][ct] = wmma_bf16(a, bb[ct], lat[rt][ct]);
    }
  }
#pragma unroll
  for (int rt = 0; rt < 4; ++rt)
#pragma unroll
    for (int ct = 0; ct < 4; ++ct) {
      int col = ct * 16 + (lane & 15);
#pragma unroll
      for (int e = 0; e < 8; ++e) {
        int row = rt * 16 + ((lane >> 4) << 3) + e;
        atomicAdd(latent + (size_t)bh * M_ * D_ + row * D_ + col, lat[rt][ct][e]);
      }
    }
}

// ------------------------------- normalize latent, store transposed bf16 [d][m]
__global__ void k_latnorm(const float* __restrict__ lat, const float* __restrict__ lsum,
                          bf16* __restrict__ latT) {
  int i = blockIdx.x * 256 + threadIdx.x;
  if (i >= B_ * H_ * M_ * D_) return;
  int d = i & 63, m = (i >> 6) & 63, bh = i >> 12;
  latT[((size_t)bh << 12) + d * M_ + m] = (bf16)(lat[i] / lsum[(bh << 6) + m]);
}

// ------------------- decode + cross attention: softmax over 64, gated combine
__device__ __forceinline__ void attn64(const bf16* Xt, const bf16* Wb, const bf16* Lb,
                                       bf16* Ps, int w, int lane, v8f (&out)[2][4]) {
  v8f lg[2][4];
#pragma unroll
  for (int rt = 0; rt < 2; ++rt)
#pragma unroll
    for (int ct = 0; ct < 4; ++ct) lg[rt][ct] = vzero8();
#pragma unroll
  for (int k0 = 0; k0 < 64; k0 += 32) {
    v16bf bb[4];
#pragma unroll
    for (int ct = 0; ct < 4; ++ct) bb[ct] = ldsB(Wb, D_, ct * 16, k0, lane);
#pragma unroll
    for (int rt = 0; rt < 2; ++rt) {
      v16bf a = ldsA(Xt, D_, w * 32 + rt * 16, k0, lane);
#pragma unroll
      for (int ct = 0; ct < 4; ++ct) lg[rt][ct] = wmma_bf16(a, bb[ct], lg[rt][ct]);
    }
  }
#pragma unroll
  for (int rt = 0; rt < 2; ++rt)
#pragma unroll
    for (int e = 0; e < 8; ++e) {
      float z0 = lg[rt][0][e], z1 = lg[rt][1][e], z2 = lg[rt][2][e], z3 = lg[rt][3][e];
      float mx = fmaxf(fmaxf(z0, z1), fmaxf(z2, z3));
#pragma unroll
      for (int m = 8; m >= 1; m >>= 1) mx = fmaxf(mx, __shfl_xor(mx, m, 32));
      float p0 = __expf(z0 - mx), p1 = __expf(z1 - mx), p2 = __expf(z2 - mx), p3 = __expf(z3 - mx);
      float s = p0 + p1 + p2 + p3;
#pragma unroll
      for (int m = 8; m >= 1; m >>= 1) s += __shfl_xor(s, m, 32);
      float inv = 1.f / s;
      int row = w * 32 + rt * 16 + ((lane >> 4) << 3) + e;
      int c = lane & 15;
      Ps[row * M_ + c]      = (bf16)(p0 * inv);
      Ps[row * M_ + 16 + c] = (bf16)(p1 * inv);
      Ps[row * M_ + 32 + c] = (bf16)(p2 * inv);
      Ps[row * M_ + 48 + c] = (bf16)(p3 * inv);
    }
#pragma unroll
  for (int rt = 0; rt < 2; ++rt)
#pragma unroll
    for (int ct = 0; ct < 4; ++ct) out[rt][ct] = vzero8();
#pragma unroll
  for (int k0 = 0; k0 < 64; k0 += 32) {
    v16bf bb[4];
#pragma unroll
    for (int ct = 0; ct < 4; ++ct) bb[ct] = ldsB(Lb, M_, ct * 16, k0, lane);
#pragma unroll
    for (int rt = 0; rt < 2; ++rt) {
      v16bf a = ldsA(Ps, M_, w * 32 + rt * 16, k0, lane);
#pragma unroll
      for (int ct = 0; ct < 4; ++ct) out[rt][ct] = wmma_bf16(a, bb[ct], out[rt][ct]);
    }
  }
}

__global__ void __launch_bounds__(128)
k_dec(const bf16* __restrict__ kbf, const bf16* __restrict__ qbf, const bf16* __restrict__ qg,
      const bf16* __restrict__ kc, const bf16* __restrict__ latT, const bf16* __restrict__ vcT,
      const float* __restrict__ state, bf16* __restrict__ omid) {
  __shared__ bf16 Xt[128 * D_];
  __shared__ bf16 Wb[M_ * D_];
  __shared__ bf16 Lb[D_ * M_];
  __shared__ bf16 Ps[128 * M_];
  const int tid = threadIdx.x, lane = tid & 31, w = tid >> 5;
  const int t0 = blockIdx.x * 128, h = blockIdx.y;
  const int b = t0 / N_, n0 = t0 - b * N_;
  const int bh = b * H_ + h;
  const size_t kbase = ((size_t)bh * N_ + n0) * D_;

  for (int i = tid; i < 1024; i += 128) async_cp16(kbf + kbase + 8 * i, Xt + 8 * i);
  {
    const bf16* g1 = qg + (size_t)h * M_ * D_;
    const bf16* g2 = latT + ((size_t)bh << 12);
    for (int i = tid; i < 512; i += 128) async_cp16(g1 + 8 * i, Wb + 8 * i);
    for (int i = tid; i < 512; i += 128) async_cp16(g2 + 8 * i, Lb + 8 * i);
  }
  wait_async0();
  __syncthreads();
  v8f sacc[2][4];
  attn64(Xt, Wb, Lb, Ps, w, lane, sacc);
  __syncthreads();
  for (int i = tid; i < 1024; i += 128) async_cp16(qbf + kbase + 8 * i, Xt + 8 * i);
  {
    const bf16* g1 = kc + ((size_t)bh << 12);
    const bf16* g2 = vcT + ((size_t)bh << 12);
    for (int i = tid; i < 512; i += 128) async_cp16(g1 + 8 * i, Wb + 8 * i);
    for (int i = tid; i < 512; i += 128) async_cp16(g2 + 8 * i, Lb + 8 * i);
  }
  wait_async0();
  __syncthreads();
  v8f cacc[2][4];
  attn64(Xt, Wb, Lb, Ps, w, lane, cacc);

  float wgt = 1.f / (1.f + __expf(-state[0]));
#pragma unroll
  for (int rt = 0; rt < 2; ++rt)
#pragma unroll
    for (int ct = 0; ct < 4; ++ct) {
      int col = ct * 16 + (lane & 15);
#pragma unroll
      for (int e = 0; e < 8; ++e) {
        int row = w * 32 + rt * 16 + ((lane >> 4) << 3) + e;
        int t = t0 + row;
        omid[(size_t)t * DIM_ + h * 64 + col] =
            (bf16)(wgt * sacc[rt][ct][e] + (1.f - wgt) * cacc[rt][ct][e]);
      }
    }
}

// ------------- final projection out_mid @ Wo^T : double-buffered async pipeline
__global__ void __launch_bounds__(128)
k_out(const bf16* __restrict__ omid, const bf16* __restrict__ Wo, const float* __restrict__ bo,
      float* __restrict__ out) {
  __shared__ bf16 As[2][128 * 32];
  __shared__ bf16 Bs[2][64 * 32];
  const int tid = threadIdx.x, lane = tid & 31, w = tid >> 5;
  const int t0 = blockIdx.x * 128, c0 = blockIdx.y * 64;

  const int r = tid >> 1, cb = (tid & 1) * 16;
  const bf16* gA = omid + (size_t)(t0 + tid) * DIM_;
  const bf16* gB = Wo + (size_t)(c0 + r) * DIM_ + cb;

  // issue one K-tile's loads: per wave 4 (A) + 2 (B) = 6 async ops
  auto issue = [&](int k0, int buf) {
    bf16* la = As[buf] + tid * 32;
#pragma unroll
    for (int i = 0; i < 4; ++i) async_cp16(gA + k0 + 8 * i, la + 8 * i);
    bf16* lb = Bs[buf] + r * 32 + cb;
    async_cp16(gB + k0, lb);
    async_cp16(gB + k0 + 8, lb + 8);
  };

  v8f acc[2][4];
#pragma unroll
  for (int rt = 0; rt < 2; ++rt)
#pragma unroll
    for (int ct = 0; ct < 4; ++ct) acc[rt][ct] = vzero8();

  issue(0, 0);
  for (int it = 0; it < DIM_ / 32; ++it) {
    int buf = it & 1;
    if (it + 1 < DIM_ / 32) {
      issue((it + 1) * 32, buf ^ 1);
      wait_async6();  // drain current tile's 6 ops (async ops complete in order)
    } else {
      wait_async0();
    }
    __syncthreads();
    v16bf bb[4];
#pragma unroll
    for (int ct = 0; ct < 4; ++ct) bb[ct] = ldsB(Bs[buf], 32, ct * 16, 0, lane);
#pragma unroll
    for (int rt = 0; rt < 2; ++rt) {
      v16bf a = ldsA(As[buf], 32, w * 32 + rt * 16, 0, lane);
#pragma unroll
      for (int ct = 0; ct < 4; ++ct) acc[rt][ct] = wmma_bf16(a, bb[ct], acc[rt][ct]);
    }
    __syncthreads();
  }
#pragma unroll
  for (int rt = 0; rt < 2; ++rt)
#pragma unroll
    for (int ct = 0; ct < 4; ++ct) {
      int col = c0 + ct * 16 + (lane & 15);
      float bc = bo[col];
#pragma unroll
      for (int e = 0; e < 8; ++e) {
        int row = w * 32 + rt * 16 + ((lane >> 4) << 3) + e;
        out[(size_t)(t0 + row) * DIM_ + col] = acc[rt][ct][e] + bc;
      }
    }
}

// =============================================================================
extern "C" void kernel_launch(void* const* d_in, const int* in_sizes, int n_in,
                              void* d_out, int out_size, void* d_ws, size_t ws_size,
                              hipStream_t stream) {
  (void)in_sizes; (void)n_in; (void)out_size; (void)ws_size;
  const float* x   = (const float*)d_in[0];
  const float* ctx = (const float*)d_in[1];
  const float* qgf = (const float*)d_in[2];
  const float* Wx  = (const float*)d_in[3];
  const float* bx  = (const float*)d_in[4];
  const float* Wk  = (const float*)d_in[5];
  const float* bk  = (const float*)d_in[6];
  const float* Wv  = (const float*)d_in[7];
  const float* bv  = (const float*)d_in[8];
  const float* Wcq = (const float*)d_in[9];
  const float* bcq = (const float*)d_in[10];
  const float* Wck = (const float*)d_in[11];
  const float* bck = (const float*)d_in[12];
  const float* Wcv = (const float*)d_in[13];
  const float* bcv = (const float*)d_in[14];
  const float* stt = (const float*)d_in[15];
  const float* Wo  = (const float*)d_in[16];
  const float* bo  = (const float*)d_in[17];
  float* out = (float*)d_out;

  char* ws = (char*)d_ws;
  size_t off = 0;
  auto take = [&](size_t bytes) -> char* {
    char* p = ws + off;
    off = (off + bytes + 255) & ~(size_t)255;
    return p;
  };
  bf16* WxB  = (bf16*)take((size_t)DIM_ * DIM_ * 2);
  bf16* WoB  = (bf16*)take((size_t)DIM_ * DIM_ * 2);
  bf16* WkB  = (bf16*)take(64 * 64 * 2);
  bf16* WvB  = (bf16*)take(64 * 64 * 2);
  bf16* WcqB = (bf16*)take(64 * 64 * 2);
  bf16* qgB  = (bf16*)take((size_t)H_ * M_ * D_ * 2);
  bf16* kcB  = (bf16*)take((size_t)B_ * H_ * SC_ * D_ * 2);
  bf16* vcTB = (bf16*)take((size_t)B_ * H_ * SC_ * D_ * 2);
  bf16* latT = (bf16*)take((size_t)B_ * H_ * M_ * D_ * 2);
  float* rmax = (float*)take((size_t)B_ * H_ * M_ * 4);
  float* lsum = (float*)take((size_t)B_ * H_ * M_ * 4);
  float* lat  = (float*)take((size_t)B_ * H_ * M_ * D_ * 4);
  bf16* kB = (bf16*)take((size_t)B_ * H_ * N_ * D_ * 2);
  bf16* qB = (bf16*)take((size_t)B_ * H_ * N_ * D_ * 2);
  bf16* vB = (bf16*)take((size_t)B_ * H_ * N_ * D_ * 2);
  bf16* omid = vB;  // v is dead after k_encacc; reuse its 32 MB for out_mid

  auto cdiv = [](int a, int b) { return (a + b - 1) / b; };

  k_f2bf<<<cdiv(DIM_ * DIM_, 256), 256, 0, stream>>>(Wx, WxB, DIM_ * DIM_);
  k_f2bf<<<cdiv(DIM_ * DIM_, 256), 256, 0, stream>>>(Wo, WoB, DIM_ * DIM_);
  k_f2bf<<<cdiv(4096, 256), 256, 0, stream>>>(Wk, WkB, 4096);
  k_f2bf<<<cdiv(4096, 256), 256, 0, stream>>>(Wv, WvB, 4096);
  k_f2bf<<<cdiv(4096, 256), 256, 0, stream>>>(Wcq, WcqB, 4096);
  k_f2bf<<<cdiv(H_ * M_ * D_, 256), 256, 0, stream>>>(qgf, qgB, H_ * M_ * D_);
  k_ctx<<<cdiv(B_ * H_ * SC_ * D_, 256), 256, 0, stream>>>(ctx, Wck, bck, Wcv, bcv, kcB, vcTB);
  k_proj_qkv<<<dim3(B_ * N_ / 128, H_), 128, 0, stream>>>(x, WxB, bx, WkB, bk, WvB, bv, WcqB, bcq,
                                                          kB, vB, qB);
  k_einit<<<cdiv(B_ * H_ * M_ * D_ + 2 * B_ * H_ * M_, 256), 256, 0, stream>>>(lat, rmax, lsum);
  k_encmax<<<dim3(B_ * H_, N_ / 128), 128, 0, stream>>>(qgB, kB, rmax);
  k_encacc<<<dim3(B_ * H_, N_ / 128), 128, 0, stream>>>(qgB, kB, vB, rmax, lsum, lat);
  k_latnorm<<<cdiv(B_ * H_ * M_ * D_, 256), 256, 0, stream>>>(lat, lsum, latT);
  k_dec<<<dim3(B_ * N_ / 128, H_), 128, 0, stream>>>(kB, qB, qgB, kcB, latT, vcTB, stt, omid);
  k_out<<<dim3(B_ * N_ / 128, DIM_ / 64), 128, 0, stream>>>(omid, WoB, bo, out);
}
